// ShiftedWindowMSA_5695126635001
// MI455X (gfx1250) — matrix operand order repaired and checked by reading, not compiled, not run on hardware
//
#include <hip/hip_runtime.h>

typedef __attribute__((ext_vector_type(16))) __bf16 v16bf;
typedef __attribute__((ext_vector_type(8)))  __bf16 v8bf;
typedef __attribute__((ext_vector_type(8)))  float  v8f;

#define B_SZ 64
#define SEQ  3136
#define ROWS (B_SZ * SEQ)   // 200704

union BF16Frag { v8bf h[2]; v16bf v; };

// A fragment: 16x32 bf16, M x K, storage row-major [M][...], ldm in elements.
// element e -> K = e + 8*(e/8) + 8*half, row M = lane%16
__device__ __forceinline__ v16bf frag_a(const __bf16* base, int ldm, int lane) {
    const int r = lane & 15, hf = lane >> 4;
    const __bf16* p = base + r * ldm + hf * 8;
    BF16Frag u;
    u.h[0] = *(const v8bf*)(p);        // K = 8*hf + 0..7
    u.h[1] = *(const v8bf*)(p + 16);   // K = 8*hf + 16..23
    return u.v;
}

// B fragment: 32x16 bf16, K x N, from N-major storage [N][K], ldn in elements.
// element e -> K = e + 8*(e/8) + 8*half, col N = lane%16
__device__ __forceinline__ v16bf frag_b(const __bf16* base, int ldn, int lane) {
    const int r = lane & 15, hf = lane >> 4;
    const __bf16* p = base + r * ldn + hf * 8;
    BF16Frag u;
    u.h[0] = *(const v8bf*)(p);        // K = 8*hf + 0..7
    u.h[1] = *(const v8bf*)(p + 16);   // K = 8*hf + 16..23
    return u.v;
}

__device__ __forceinline__ v8f wmma_bf16(v16bf a, v16bf b, v8f c) {
    return __builtin_amdgcn_wmma_f32_16x16x32_bf16(false, a, false, b, (short)0, c, false, false);
}

// ---------------- Kernel 1: qkv = x @ w1.T + b1 (bf16 out, channels de-interleaved) -----------
__global__ void __launch_bounds__(256) qkv_gemm_kernel(const float* __restrict__ x,
                                                       const float* __restrict__ w1,
                                                       const float* __restrict__ b1,
                                                       __bf16* __restrict__ qkv) {
    __shared__ __align__(16) __bf16 Xs[128][96];
    __shared__ __align__(16) __bf16 Ws[96][96];
    const int tid = threadIdx.x, lane = tid & 31, wv = tid >> 5;
    const long rowBase = (long)blockIdx.x * 128;

    for (int i = tid; i < 128 * 96; i += 256) {
        int r = i / 96, k = i - r * 96;
        Xs[r][k] = (__bf16)x[(rowBase + r) * 96 + k];
    }
    __syncthreads();

    for (int chunk = 0; chunk < 3; ++chunk) {
        if (chunk) __syncthreads();
        for (int i = tid; i < 96 * 96; i += 256) {
            int n = i / 96, k = i - n * 96;
            Ws[n][k] = (__bf16)w1[(chunk * 96 + n) * 96 + k];
        }
        __syncthreads();

        v8f acc[6] = {};
#pragma unroll
        for (int kc = 0; kc < 3; ++kc) {
            v16bf a = frag_a(&Xs[wv * 16][kc * 32], 96, lane);
#pragma unroll
            for (int nt = 0; nt < 6; ++nt) {
                v16bf bb = frag_b(&Ws[nt * 16][kc * 32], 96, lane);
                acc[nt] = wmma_bf16(a, bb, acc[nt]);
            }
        }
        const int r = lane & 15, hf = lane >> 4;
#pragma unroll
        for (int nt = 0; nt < 6; ++nt) {
            int norig = chunk * 96 + nt * 16 + r;
            int kk = norig % 3, e = norig / 3;     // (E,3) split, 3 fastest
            float bv = b1[norig];
#pragma unroll
            for (int v = 0; v < 8; ++v) {
                long row = rowBase + wv * 16 + v + 8 * hf;
                qkv[row * 288 + kk * 96 + e] = (__bf16)(acc[nt][v] + bv);
            }
        }
    }
}

// ---------------- Kernel 2: shifted-window attention -----------------------------------------
__global__ void __launch_bounds__(384) attn_kernel(const __bf16* __restrict__ qkvp,
                                                   __bf16* __restrict__ osc) {
    __shared__ __align__(16) __bf16 Qs[3][64][32];
    __shared__ __align__(16) __bf16 Ks[3][64][32];
    __shared__ __align__(16) __bf16 Vt[3][32][64];   // V transposed: [d][tok]
    __shared__ __align__(16) float  Sb[12][16][64];
    __shared__ __align__(16) __bf16 At[12][16][64];

    const int tid = threadIdx.x, lane = tid & 31, wv = tid >> 5;
    const int hd = wv >> 2, mt = wv & 3;
    const int b  = blockIdx.x >> 6, win = blockIdx.x & 63;
    const int iy = win >> 3, ix = win & 7;

    {   // zero staging (covers token pads 49..63)
        unsigned int* z0 = (unsigned int*)&Qs[0][0][0];
        unsigned int* z1 = (unsigned int*)&Ks[0][0][0];
        unsigned int* z2 = (unsigned int*)&Vt[0][0][0];
        for (int i = tid; i < 3072; i += 384) { z0[i] = 0u; z1[i] = 0u; z2[i] = 0u; }
    }
    __syncthreads();

    // gather window tokens with forward roll (+4)
    const unsigned int* src = (const unsigned int*)qkvp;
    for (int w = tid; w < 49 * 144; w += 384) {
        int tok = w / 144, widx = w - tok * 144;
        int m1 = tok / 7, m2 = tok - m1 * 7;
        int y  = iy * 7 + m1 + 4; if (y  >= 56) y  -= 56;
        int xx = ix * 7 + m2 + 4; if (xx >= 56) xx -= 56;
        long s = (long)b * SEQ + y * 56 + xx;
        unsigned int val = src[s * 144 + widx];
        int kk = widx / 48, rem = widx - kk * 48;
        int e0 = rem * 2, h2 = e0 >> 5, d0 = e0 & 31;
        if (kk == 0)      ((unsigned int*)&Qs[h2][tok][0])[d0 >> 1] = val;
        else if (kk == 1) ((unsigned int*)&Ks[h2][tok][0])[d0 >> 1] = val;
        else {
            union { unsigned int u; __bf16 bh[2]; } cv; cv.u = val;
            Vt[h2][d0][tok]     = cv.bh[0];
            Vt[h2][d0 + 1][tok] = cv.bh[1];
        }
    }
    __syncthreads();

    // S = Q K^T
    v8f s[4] = {};
    {
        v16bf a = frag_a(&Qs[hd][mt * 16][0], 32, lane);
#pragma unroll
        for (int nt = 0; nt < 4; ++nt) {
            v16bf bb = frag_b(&Ks[hd][nt * 16][0], 32, lane);
            s[nt] = wmma_bf16(a, bb, s[nt]);
        }
    }
    const float scale = 0.17677669529663687f;   // 1/sqrt(32)
    const int r = lane & 15, hf = lane >> 4;
#pragma unroll
    for (int nt = 0; nt < 4; ++nt)
#pragma unroll
        for (int v = 0; v < 8; ++v)
            Sb[wv][v + 8 * hf][nt * 16 + r] = s[nt][v] * scale;

    // masked softmax per row (within-wave; LDS ops are in-order per wave)
    if (lane < 16) {
        int tm = mt * 16 + lane;
        if (tm < 49) {
            const bool lastR = (iy == 7), lastC = (ix == 7);
            const bool mr = (tm / 7) >= 4;
            const bool mc = (tm % 7) >= 4;
            float mx = -3.0e38f;
            for (int n = 0; n < 49; ++n) {
                bool msk = (lastR && (((n / 7) >= 4) != mr)) || (lastC && (((n % 7) >= 4) != mc));
                float sv = Sb[wv][lane][n];
                if (!msk && sv > mx) mx = sv;
            }
            float sum = 0.f;
            for (int n = 0; n < 49; ++n) {
                bool msk = (lastR && (((n / 7) >= 4) != mr)) || (lastC && (((n % 7) >= 4) != mc));
                float ev = msk ? 0.f : __expf(Sb[wv][lane][n] - mx);
                Sb[wv][lane][n] = ev;
                sum += ev;
            }
            float inv = 1.f / sum;
            for (int n = 0; n < 64; ++n)
                At[wv][lane][n] = (__bf16)((n < 49) ? Sb[wv][lane][n] * inv : 0.f);
        } else {
            for (int n = 0; n < 64; ++n) At[wv][lane][n] = (__bf16)0.f;
        }
    }

    // O = att @ V  (K padded to 64, att pad cols are zero)
    v8f o[2] = {};
#pragma unroll
    for (int kc = 0; kc < 2; ++kc) {
        v16bf aa = frag_a(&At[wv][0][kc * 32], 64, lane);
#pragma unroll
        for (int nt = 0; nt < 2; ++nt) {
            v16bf bb = frag_b(&Vt[hd][nt * 16][kc * 32], 64, lane);
            o[nt] = wmma_bf16(aa, bb, o[nt]);
        }
    }

    // scatter with inverse roll (+3)
#pragma unroll
    for (int nt = 0; nt < 2; ++nt)
#pragma unroll
        for (int v = 0; v < 8; ++v) {
            int tm = mt * 16 + v + 8 * hf;
            if (tm < 49) {
                int m1 = tm / 7, m2 = tm - m1 * 7;
                int y  = iy * 7 + m1 + 3; if (y  >= 56) y  -= 56;
                int xx = ix * 7 + m2 + 3; if (xx >= 56) xx -= 56;
                long sd = (long)b * SEQ + y * 56 + xx;
                osc[sd * 96 + hd * 32 + nt * 16 + r] = (__bf16)o[nt][v];
            }
        }
}

// ---------------- Kernel 3: out = o @ w2.T + b2 (f32 out) -------------------------------------
__global__ void __launch_bounds__(256) proj_gemm_kernel(const __bf16* __restrict__ oin,
                                                        const float* __restrict__ w2,
                                                        const float* __restrict__ b2,
                                                        float* __restrict__ out) {
    __shared__ __align__(16) __bf16 Xs[128][96];
    __shared__ __align__(16) __bf16 Ws[96][96];
    const int tid = threadIdx.x, lane = tid & 31, wv = tid >> 5;
    const long rowBase = (long)blockIdx.x * 128;

    const unsigned int* srcw = (const unsigned int*)oin;
    unsigned int* xw = (unsigned int*)&Xs[0][0];
    for (int i = tid; i < 128 * 48; i += 256) xw[i] = srcw[rowBase * 48 + i];
    for (int i = tid; i < 96 * 96; i += 256) {
        int n = i / 96, k = i - n * 96;
        Ws[n][k] = (__bf16)w2[i];
    }
    __syncthreads();

    v8f acc[6] = {};
#pragma unroll
    for (int kc = 0; kc < 3; ++kc) {
        v16bf a = frag_a(&Xs[wv * 16][kc * 32], 96, lane);
#pragma unroll
        for (int nt = 0; nt < 6; ++nt) {
            v16bf bb = frag_b(&Ws[nt * 16][kc * 32], 96, lane);
            acc[nt] = wmma_bf16(a, bb, acc[nt]);
        }
    }
    const int r = lane & 15, hf = lane >> 4;
#pragma unroll
    for (int nt = 0; nt < 6; ++nt) {
        int n = nt * 16 + r;
        float bv = b2[n];
#pragma unroll
        for (int v = 0; v < 8; ++v) {
            long row = rowBase + wv * 16 + v + 8 * hf;
            out[row * 96 + n] = acc[nt][v] + bv;
        }
    }
}

extern "C" void kernel_launch(void* const* d_in, const int* in_sizes, int n_in,
                              void* d_out, int out_size, void* d_ws, size_t ws_size,
                              hipStream_t stream) {
    const float* x  = (const float*)d_in[0];
    const float* w1 = (const float*)d_in[1];
    const float* b1 = (const float*)d_in[2];
    const float* w2 = (const float*)d_in[3];
    const float* b2 = (const float*)d_in[4];
    float* out = (float*)d_out;

    char* ws = (char*)d_ws;
    __bf16* qkv = (__bf16*)ws;                                  // 200704*288*2 = 115,605,504 B
    __bf16* osc = (__bf16*)(ws + (size_t)ROWS * 288 * 2);       // 200704*96*2  =  38,535,168 B

    qkv_gemm_kernel<<<ROWS / 128, 256, 0, stream>>>(x, w1, b1, qkv);
    attn_kernel<<<B_SZ * 64, 384, 0, stream>>>(qkv, osc);
    proj_gemm_kernel<<<ROWS / 128, 256, 0, stream>>>(osc, w2, b2, out);
}
